// GroupedQueryAttention_6966436954226
// MI455X (gfx1250) — matrix operand order repaired and checked by reading, compiled, not run on hardware
//
#include <hip/hip_runtime.h>

// ---------- types ----------
typedef __attribute__((ext_vector_type(16))) __bf16        v16bf;
typedef __attribute__((ext_vector_type(8)))  float         v8f;
typedef __attribute__((ext_vector_type(4)))  unsigned int  u32x4;
typedef __attribute__((ext_vector_type(4)))  float         f32x4;

// exact parameter type of the async-to-LDS builtins (from hipcc diagnostic):
// '__attribute__((__vector_size__(4 * sizeof(int)))) int __device__ *'
typedef int v4i __attribute__((vector_size(16)));
typedef __attribute__((address_space(1))) v4i* gv4i_p;   // global (AS1)
typedef __attribute__((address_space(3))) v4i* lv4i_p;   // LDS (AS3)

union BFrag {
    v16bf        v;
    u32x4        q[2];
    unsigned int u[8];
};

// CDNA5 async global->LDS copy path (ASYNCcnt), with portable fallback.
#if defined(__has_builtin)
#if __has_builtin(__builtin_amdgcn_global_load_async_to_lds_b128) && \
    __has_builtin(__builtin_amdgcn_s_wait_asynccnt)
#define HAVE_ASYNC 1
#endif
#endif
#ifndef HAVE_ASYNC
#define HAVE_ASYNC 0
#endif

__device__ __forceinline__ unsigned short f2bf(float f) {
    unsigned int u = __float_as_uint(f);
    u += 0x7FFFu + ((u >> 16) & 1u);           // round-to-nearest-even
    return (unsigned short)(u >> 16);
}

__device__ __forceinline__ void async_cp16(const unsigned short* src, unsigned short* dst) {
#if HAVE_ASYNC
    __builtin_amdgcn_global_load_async_to_lds_b128(
        (gv4i_p)(v4i*)(unsigned short*)src,
        (lv4i_p)(v4i*)dst, 0, 0);
#else
    *(u32x4*)dst = *(const u32x4*)src;
#endif
}
__device__ __forceinline__ void async_join() {
#if HAVE_ASYNC
    __builtin_amdgcn_s_wait_asynccnt(0);
#endif
}

// =====================================================================
// One-time conversion passes (O(N^2), bandwidth bound)
// =====================================================================
__global__ __launch_bounds__(256)
void cvt_bf16_kernel(const float* __restrict__ in, unsigned short* __restrict__ out, int n4) {
    int i = blockIdx.x * 256 + threadIdx.x;
    if (i < n4) {
        f32x4 a = ((const f32x4*)in)[i];
        unsigned long long p =
             (unsigned long long)f2bf(a.x)
           | ((unsigned long long)f2bf(a.y) << 16)
           | ((unsigned long long)f2bf(a.z) << 32)
           | ((unsigned long long)f2bf(a.w) << 48);
        ((unsigned long long*)out)[i] = p;
    }
}

// W[K,N] f32 -> Wt[N,K] bf16 (tiled transpose, both sides coalesced)
__global__ __launch_bounds__(256)
void transpose_bf16_kernel(const float* __restrict__ W, unsigned short* __restrict__ Wt,
                           int K, int N) {
    __shared__ float tile[32][33];
    const int k0 = blockIdx.y * 32, n0 = blockIdx.x * 32;
    const int tx = threadIdx.x & 31, ty = threadIdx.x >> 5;   // 32 x 8
    #pragma unroll
    for (int i = 0; i < 32; i += 8)
        tile[ty + i][tx] = W[(size_t)(k0 + ty + i) * N + n0 + tx];
    __syncthreads();
    #pragma unroll
    for (int i = 0; i < 32; i += 8)
        Wt[(size_t)(n0 + ty + i) * K + k0 + tx] = f2bf(tile[tx][ty + i]);
}

// =====================================================================
// GEMM: C[M,N] = A[M,K](bf16) * Wt[N,K](bf16, pre-transposed) + bias[N]
// 128x128 block tile, K-step 32, 8 waves, double-buffered async staging.
// =====================================================================
__device__ __forceinline__ void stage_128x32(const unsigned short* g0, size_t stride,
                                             int k0, unsigned short* lds, int t) {
    #pragma unroll
    for (int i = 0; i < 2; ++i) {              // 512 x 16B chunks / 256 thr
        int idx = i * 256 + t;
        int row = idx >> 2, c4 = idx & 3;
        async_cp16(g0 + (size_t)row * stride + k0 + c4 * 8, lds + row * 32 + c4 * 8);
    }
}

template <bool BF16OUT>
__global__ __launch_bounds__(256)
void gemm_bf16_wmma(const unsigned short* __restrict__ A,   // [M,K] bf16
                    const unsigned short* __restrict__ Wt,  // [N,K] bf16
                    const float* __restrict__ bias, void* __restrict__ Cout,
                    int M, int N, int K)
{
    __shared__ unsigned short As[2][128 * 32];
    __shared__ unsigned short Bs[2][128 * 32];

    const int t    = threadIdx.x;
    const int wv   = t >> 5;
    const int lane = t & 31;
    const int lq   = lane & 15;
    const int half = lane >> 4;
    const int wm   = wv >> 1;                  // 0..3 -> 32 rows
    const int wn   = wv & 1;                   // 0..1 -> 64 cols

    const int m0 = blockIdx.y * 128;
    const int n0 = blockIdx.x * 128;
    const unsigned short* Ag = A  + (size_t)m0 * K;
    const unsigned short* Bg = Wt + (size_t)n0 * K;

    v8f acc[2][4] = {};

    const int KT = K >> 5;
    stage_128x32(Ag, K, 0, As[0], t);
    stage_128x32(Bg, K, 0, Bs[0], t);

    for (int kt = 0; kt < KT; ++kt) {
        const int cur = kt & 1;
        async_join();
        __syncthreads();                       // buf[cur] ready; buf[cur^1] free
        if (kt + 1 < KT) {
            stage_128x32(Ag, K, (kt + 1) * 32, As[cur ^ 1], t);
            stage_128x32(Bg, K, (kt + 1) * 32, Bs[cur ^ 1], t);
        }
        const unsigned short* Asc = As[cur];
        const unsigned short* Bsc = Bs[cur];

        BFrag afr[2], bfr[4];
        #pragma unroll
        for (int im = 0; im < 2; ++im) {       // A frag 16x32 (ISA layout)
            int row = wm * 32 + im * 16 + lq;
            afr[im].q[0] = *(const u32x4*)&Asc[row * 32 + half * 8];
            afr[im].q[1] = *(const u32x4*)&Asc[row * 32 + 16 + half * 8];
        }
        #pragma unroll
        for (int in = 0; in < 4; ++in) {       // B frag 32x16 from [n][k] tile
            int n = wn * 64 + in * 16 + lq;
            bfr[in].q[0] = *(const u32x4*)&Bsc[n * 32 + half * 8];
            bfr[in].q[1] = *(const u32x4*)&Bsc[n * 32 + 16 + half * 8];
        }
        #pragma unroll
        for (int im = 0; im < 2; ++im)
            #pragma unroll
            for (int in = 0; in < 4; ++in)
                acc[im][in] = __builtin_amdgcn_wmma_f32_16x16x32_bf16(
                    false, afr[im].v, false, bfr[in].v, (short)0, acc[im][in], false, false);
    }

    #pragma unroll
    for (int in = 0; in < 4; ++in) {
        int n = n0 + wn * 64 + in * 16 + lq;
        float bvv = bias[n];
        #pragma unroll
        for (int im = 0; im < 2; ++im)
            #pragma unroll
            for (int r = 0; r < 8; ++r) {
                int m = m0 + wm * 32 + im * 16 + r + half * 8;
                float val = acc[im][in][r] + bvv;
                if constexpr (BF16OUT)
                    ((unsigned short*)Cout)[(size_t)m * N + n] = f2bf(val);
                else
                    ((float*)Cout)[(size_t)m * N + n] = val;
            }
    }
}

// =====================================================================
// Flash GQA attention. Block = (b, query head, 64 queries), 4 waves.
// S^T = K*Q^T (query = lane dim -> cheap softmax); ctx = P*V with P as
// A operand (in-lane repack, zero shuffles); coalesced bf16 ctx stores.
// =====================================================================
#define SDIM   2048
#define DMODEL 2048
#define NKV    8
#define NH     32
#define DHD    64

__global__ __launch_bounds__(128)
void gqa_flash_kernel(const unsigned short* __restrict__ Qb,   // bf16 [B,S,D]
                      const unsigned short* __restrict__ Kb,   // bf16 [B,S,NKV*DH]
                      const unsigned short* __restrict__ Vb,   // bf16 [B,S,NKV*DH]
                      unsigned short* __restrict__ Out)        // bf16 [B,S,D] merged
{
    __shared__ unsigned short Kl[64 * 64];     // [key][d]
    __shared__ unsigned short Vt[64 * 64];     // [d][key]

    const int t    = threadIdx.x;
    const int wv   = t >> 5;
    const int lane = t & 31;
    const int lq   = lane & 15;
    const int half = lane >> 4;

    const int QB   = SDIM / 64;
    const int bid  = blockIdx.x;
    const int b    = bid / (NH * QB);
    const int rem  = bid % (NH * QB);
    const int hq   = rem / QB;
    const int qblk = rem % QB;
    const int hkv  = hq / (NH / NKV);

    const int qbase = qblk * 64 + wv * 16;
    const int qg    = qbase + lq;              // this lane's query (stats owner)

    BFrag qf[2];                               // Q^T B-fragments, loop invariant
    {
        const unsigned short* qp = Qb + ((size_t)(b * SDIM + qg)) * DMODEL + hq * DHD;
        #pragma unroll
        for (int c = 0; c < 2; ++c) {
            qf[c].q[0] = *(const u32x4*)(qp + c * 32 + half * 8);
            qf[c].q[1] = *(const u32x4*)(qp + c * 32 + 16 + half * 8);
        }
    }

    v8f ctx[4] = {};                           // 16q x 64d (q rows, d lanes)
    float mrun = -3.0e38f, lrun = 0.0f;

    for (int kb = 0; kb < SDIM; kb += 64) {
        __syncthreads();
        // K tile: pure bf16 copy -> async-to-LDS
        #pragma unroll
        for (int i = 0; i < 4; ++i) {
            int idx = i * 128 + t;             // 512 x 16B
            int key = idx >> 3, d8 = idx & 7;
            async_cp16(Kb + ((size_t)(b * SDIM + kb + key)) * (NKV * DHD) + hkv * DHD + d8 * 8,
                       &Kl[key * 64 + d8 * 8]);
        }
        // V transposed: pack two keys per b32 store
        #pragma unroll
        for (int i = 0; i < 16; ++i) {
            int idx = i * 128 + t;             // 2048 key-pairs
            int d = idx & 63, j = idx >> 6;    // j = key pair 0..31
            const unsigned short* vp =
                Vb + ((size_t)(b * SDIM + kb + 2 * j)) * (NKV * DHD) + hkv * DHD + d;
            unsigned int p = (unsigned int)vp[0] | ((unsigned int)vp[NKV * DHD] << 16);
            *(unsigned int*)&Vt[d * 64 + 2 * j] = p;
        }
        async_join();
        __syncthreads();

        // ---- S^T: 4 tiles of 16 keys x 16 queries ----
        v8f s[4];
        #pragma unroll
        for (int kt = 0; kt < 4; ++kt) {
            v8f a = {};
            #pragma unroll
            for (int c = 0; c < 2; ++c) {
                BFrag kf;
                int row = kt * 16 + lq;
                kf.q[0] = *(const u32x4*)&Kl[row * 64 + c * 32 + half * 8];
                kf.q[1] = *(const u32x4*)&Kl[row * 64 + c * 32 + 16 + half * 8];
                a = __builtin_amdgcn_wmma_f32_16x16x32_bf16(
                        false, kf.v, false, qf[c].v, (short)0, a, false, false);
            }
            #pragma unroll
            for (int r = 0; r < 8; ++r) a[r] *= 0.125f;   // 1/sqrt(64)
            s[kt] = a;
        }

        // ---- online softmax (per lane query; xor-16 joins halves) ----
        float bm = -3.0e38f;
        #pragma unroll
        for (int kt = 0; kt < 4; ++kt)
            #pragma unroll
            for (int r = 0; r < 8; ++r) bm = fmaxf(bm, s[kt][r]);
        bm = fmaxf(bm, __shfl_xor(bm, 16, 32));
        float newm  = fmaxf(mrun, bm);
        float alpha = __expf(mrun - newm);
        mrun = newm;

        float ps = 0.0f;
        #pragma unroll
        for (int kt = 0; kt < 4; ++kt)
            #pragma unroll
            for (int r = 0; r < 8; ++r) {
                float e = __expf(s[kt][r] - mrun);
                s[kt][r] = e;
                ps += e;
            }
        ps += __shfl_xor(ps, 16, 32);
        lrun = lrun * alpha + ps;

        // rescale ctx: row r holds query (r + half*8) -> broadcast alpha per row
        #pragma unroll
        for (int r = 0; r < 8; ++r) {
            float ar = __shfl(alpha, r + half * 8, 32);
            #pragma unroll
            for (int dt = 0; dt < 4; ++dt) ctx[dt][r] *= ar;
        }

        // ---- ctx += P * V : P as A operand (pure in-lane repack) ----
        #pragma unroll
        for (int c = 0; c < 2; ++c) {
            BFrag pf;                          // A-frag element (v,e) = P[q][kk]
            #pragma unroll
            for (int v = 0; v < 8; ++v) {
                const v8f& st = s[2 * c + (v >> 2)];
                int r = 2 * (v & 3);
                pf.u[v] = (unsigned int)f2bf(st[r]) |
                          ((unsigned int)f2bf(st[r + 1]) << 16);
            }
            #pragma unroll
            for (int dt = 0; dt < 4; ++dt) {
                BFrag vf;                      // B-frag: V (32 keys x 16 d) from Vt
                int d = dt * 16 + lq;
                vf.q[0] = *(const u32x4*)&Vt[d * 64 + c * 32 + half * 8];
                vf.q[1] = *(const u32x4*)&Vt[d * 64 + c * 32 + 16 + half * 8];
                ctx[dt] = __builtin_amdgcn_wmma_f32_16x16x32_bf16(
                              false, pf.v, false, vf.v, (short)0, ctx[dt], false, false);
            }
        }
    }

    // ---- finalize: /l, coalesced bf16 stores (lane = d dimension) ----
    float rl = 1.0f / lrun;
    #pragma unroll
    for (int r = 0; r < 8; ++r) {
        float rr = __shfl(rl, r + half * 8, 32);
        int q = qbase + r + half * 8;
        #pragma unroll
        for (int dt = 0; dt < 4; ++dt)
            Out[((size_t)(b * SDIM + q)) * DMODEL + hq * DHD + dt * 16 + lq] =
                f2bf(ctx[dt][r] * rr);
    }
}

// =====================================================================
// Launch
// =====================================================================
extern "C" void kernel_launch(void* const* d_in, const int* in_sizes, int n_in,
                              void* d_out, int out_size, void* d_ws, size_t ws_size,
                              hipStream_t stream) {
    const float* x  = (const float*)d_in[0];
    const float* Wq = (const float*)d_in[1];
    const float* bq = (const float*)d_in[2];
    const float* Wk = (const float*)d_in[3];
    const float* bk = (const float*)d_in[4];
    const float* Wv = (const float*)d_in[5];
    const float* bv = (const float*)d_in[6];
    const float* Wo = (const float*)d_in[7];
    const float* bo = (const float*)d_in[8];
    float* out = (float*)d_out;

    const int Bv = 2, S = 2048, D = 2048, NKVD = 512;
    const int M = Bv * S;                       // 4096

    unsigned short* p = (unsigned short*)d_ws;
    unsigned short* xbf = p;  p += (size_t)M * D;        // x bf16
    unsigned short* WqT = p;  p += (size_t)D * D;        // [N=D, K=D]
    unsigned short* WkT = p;  p += (size_t)NKVD * D;     // [N=512, K=D]
    unsigned short* WvT = p;  p += (size_t)NKVD * D;
    unsigned short* WoT = p;  p += (size_t)D * D;
    unsigned short* qbf = p;  p += (size_t)M * D;
    unsigned short* kbf = p;  p += (size_t)M * NKVD;
    unsigned short* vbf = p;  p += (size_t)M * NKVD;
    unsigned short* mbf = p;  p += (size_t)M * D;        // merged ctx bf16

    dim3 blk(256);
    // one-time conversions
    int n4 = (int)((size_t)M * D / 4);
    cvt_bf16_kernel<<<(n4 + 255) / 256, blk, 0, stream>>>(x, xbf, n4);
    transpose_bf16_kernel<<<dim3(D    / 32, D / 32), blk, 0, stream>>>(Wq, WqT, D, D);
    transpose_bf16_kernel<<<dim3(NKVD / 32, D / 32), blk, 0, stream>>>(Wk, WkT, D, NKVD);
    transpose_bf16_kernel<<<dim3(NKVD / 32, D / 32), blk, 0, stream>>>(Wv, WvT, D, NKVD);
    transpose_bf16_kernel<<<dim3(D    / 32, D / 32), blk, 0, stream>>>(Wo, WoT, D, D);

    // projections (bf16 out)
    gemm_bf16_wmma<true ><<<dim3(D    / 128, M / 128), blk, 0, stream>>>(xbf, WqT, bq, qbf, M, D,    D);
    gemm_bf16_wmma<true ><<<dim3(NKVD / 128, M / 128), blk, 0, stream>>>(xbf, WkT, bk, kbf, M, NKVD, D);
    gemm_bf16_wmma<true ><<<dim3(NKVD / 128, M / 128), blk, 0, stream>>>(xbf, WvT, bv, vbf, M, NKVD, D);

    // attention
    gqa_flash_kernel<<<dim3(Bv * NH * (S / 64)), dim3(128), 0, stream>>>(qbf, kbf, vbf, mbf);

    // output projection (f32 out)
    gemm_bf16_wmma<false><<<dim3(D / 128, M / 128), blk, 0, stream>>>(mbf, WoT, bo, out, M, D, D);
}
